// Decoder_41747082117785
// MI455X (gfx1250) — 8-bit matrix operand order repaired and checked by reading, compiled, not run on hardware
//
#include <hip/hip_runtime.h>


typedef __attribute__((ext_vector_type(16))) int   v16i;
typedef __attribute__((ext_vector_type(8)))  float v8f;

#define BM   64
#define S8   272
#define SF   260
#define SMEM_BYTES (2*BM*S8 + BM*SF*4 + 512*4 + 2*BM*4)

union V16U {
  v16i v;
  uint4 q[4];
  unsigned long long d[8];
};

__device__ __forceinline__ unsigned char f32_to_e4m3(float x) {
  unsigned int u = __float_as_uint(x);
  unsigned int s = (u >> 24) & 0x80u;
  float a = fabsf(x);
  if (!(a < 448.0f)) return (unsigned char)(s | 0x7e);
  int E; float m = frexpf(a, &E);
  int e = E + 6;
  int code;
  if (e <= 0) {
    code = (int)(a * 512.0f + 0.5f);
  } else {
    int frac = (int)(m * 16.0f + 0.5f);
    if (frac == 16) { frac = 8; e += 1; }
    code = (e << 3) | (frac - 8);
    if (code > 0x7e) code = 0x7e;
  }
  return (unsigned char)(s | code);
}

__device__ __forceinline__ float fast_tanh(float x) {
#if __has_builtin(__builtin_amdgcn_tanhf)
  return __builtin_amdgcn_tanhf(x);
#else
  float q = __builtin_amdgcn_exp2f(x * 2.885390082f);
  return 1.0f - 2.0f * __builtin_amdgcn_rcpf(q + 1.0f);
#endif
}
__device__ __forceinline__ float fast_sigmoid(float x) {
  return 0.5f + 0.5f * fast_tanh(0.5f * x);
}

__global__ void lstm_prep(const float* __restrict__ Wih, const float* __restrict__ Whh,
                          const float* __restrict__ fcW, unsigned char* __restrict__ wfrag) {
  int idx = blockIdx.x * 256 + threadIdx.x;
  if (idx >= 64 * 2 * 32 * 64) return;
  int b    = idx & 63;
  int lane = (idx >> 6) & 31;
  int ks   = (idx >> 11) & 1;
  int ntg  = idx >> 12;
  int i = b >> 2, bb = b & 3;
  int lh = lane >> 4, ln = lane & 15;
  int K = ks * 128 + (i >> 2) * 32 + lh * 16 + (i & 3) * 4 + bb;
  int N = ntg * 16 + ln;
  float wp = Whh[N * 256 + K]
           + Wih[N * 2 + 0] * fcW[0 * 256 + K]
           + Wih[N * 2 + 1] * fcW[1 * 256 + K];
  wfrag[idx] = f32_to_e4m3(wp * 16.0f);
}

__global__ __launch_bounds__(256) void lstm_decode(
    const float* __restrict__ h0g, const float* __restrict__ c0g,
    const float* __restrict__ x0g, const float* __restrict__ Wih,
    const float* __restrict__ bih, const float* __restrict__ bhh,
    const float* __restrict__ fcW, const float* __restrict__ fcb,
    const unsigned char* __restrict__ wfrag, const int* __restrict__ Tp,
    float* __restrict__ out) {
  extern __shared__ char smem[];
  unsigned char* hbuf = (unsigned char*)smem;
  float* hf    = (float*)(smem + 2 * BM * S8);
  float* fcw_s = hf + BM * SF;
  float* dx    = fcw_s + 512;

  const int tid  = threadIdx.x;
  const int lane = tid & 31;
  const int w    = tid >> 5;
  const int lh   = lane >> 4;
  const int ln   = lane & 15;
  const int rowbase = blockIdx.x * BM;
  const int T = Tp[0];
  const float fcb0 = fcb[0], fcb1 = fcb[1];

  v16i bfrag[2][4][2];
#pragma unroll
  for (int cl = 0; cl < 2; ++cl)
#pragma unroll
    for (int g = 0; g < 4; ++g)
#pragma unroll
      for (int ks = 0; ks < 2; ++ks) {
        int ntg = g * 16 + w * 2 + cl;
        const uint4* p = (const uint4*)(wfrag + (size_t)((ntg * 2 + ks) * 32 + lane) * 64);
        V16U t; t.q[0] = p[0]; t.q[1] = p[1]; t.q[2] = p[2]; t.q[3] = p[3];
        bfrag[cl][g][ks] = t.v;
      }

  float creg[4][2][8];
#pragma unroll
  for (int mt = 0; mt < 4; ++mt)
#pragma unroll
    for (int cl = 0; cl < 2; ++cl)
#pragma unroll
      for (int r = 0; r < 8; ++r) {
        int m = mt * 16 + r + 8 * lh;
        int j = w * 32 + cl * 16 + ln;
        creg[mt][cl][r] = c0g[(size_t)(rowbase + m) * 256 + j];
      }

  float bfr[2][4];
#pragma unroll
  for (int cl = 0; cl < 2; ++cl)
#pragma unroll
    for (int g = 0; g < 4; ++g) {
      int n = g * 256 + w * 32 + cl * 16 + ln;
      bfr[cl][g] = bih[n] + bhh[n] + fcb0 * Wih[n * 2 + 0] + fcb1 * Wih[n * 2 + 1];
    }

  for (int i = tid; i < 512; i += 256) fcw_s[i] = fcW[i];
  for (int idx = tid; idx < BM * 256; idx += 256) {
    int m = idx >> 8, j = idx & 255;
    float v = h0g[(size_t)(rowbase + m) * 256 + j];
    hf[m * SF + j] = v;
    hbuf[m * S8 + j] = f32_to_e4m3(v);
  }
  __syncthreads();

  if (tid < 2 * BM) {
    int m = tid >> 1, o = tid & 1;
    const float4* hp  = (const float4*)(hf + m * SF);
    const float4* wp4 = (const float4*)(fcw_s + o * 256);
    float acc = 0.f;
#pragma unroll 8
    for (int jj = 0; jj < 64; ++jj) {
      float4 hv = hp[jj], wv = wp4[jj];
      acc += hv.x * wv.x + hv.y * wv.y + hv.z * wv.z + hv.w * wv.w;
    }
    dx[m * 2 + o] = x0g[(size_t)(rowbase + m) * 2 + o] - (acc + ((o == 0) ? fcb0 : fcb1));
  }
  __syncthreads();

  for (int t = 0; t < T; ++t) {
    unsigned char* hcur = hbuf + (t & 1) * BM * S8;
    unsigned char* hnxt = hbuf + ((t + 1) & 1) * BM * S8;

#pragma unroll
    for (int mt = 0; mt < 4; ++mt) {
      V16U af[2];
#pragma unroll
      for (int ks = 0; ks < 2; ++ks) {
        const unsigned char* rp = hcur + (mt * 16 + ln) * S8 + ks * 128 + lh * 16;
#pragma unroll
        for (int hh = 0; hh < 2; ++hh)
#pragma unroll
          for (int tq = 0; tq < 4; ++tq)
            af[ks].d[hh * 4 + tq] =
                *(const unsigned long long*)(rp + hh * 64 + (tq >> 1) * 32 + (tq & 1) * 8);
      }

      float dxp0[8], dxp1[8];
      if (t == 0) {
#pragma unroll
        for (int r = 0; r < 8; ++r) {
          int m2 = mt * 16 + r + 8 * lh;
          float2 dd = *(const float2*)(dx + m2 * 2);
          dxp0[r] = dd.x; dxp1[r] = dd.y;
        }
      }

#pragma unroll
      for (int cl = 0; cl < 2; ++cl) {
        v8f acc4[4];
#pragma unroll
        for (int g = 0; g < 4; ++g) {
          v8f a0 = {};
          a0 = __builtin_amdgcn_wmma_f32_16x16x128_fp8_fp8(
              af[0].v, bfrag[cl][g][0], (short)0, a0, false, false);
          a0 = __builtin_amdgcn_wmma_f32_16x16x128_fp8_fp8(
              af[1].v, bfrag[cl][g][1], (short)0, a0, false, false);
          acc4[g] = a0;
        }
        if (t == 0) {
#pragma unroll
          for (int g = 0; g < 4; ++g) {
            int n = g * 256 + w * 32 + cl * 16 + ln;
            float wi0 = Wih[n * 2 + 0] * 16.0f;
            float wi1 = Wih[n * 2 + 1] * 16.0f;
#pragma unroll
            for (int r = 0; r < 8; ++r)
              acc4[g][r] = acc4[g][r] + dxp0[r] * wi0 + dxp1[r] * wi1;
          }
        }
        const int j = w * 32 + cl * 16 + ln;
        float hv[8];
#pragma unroll
        for (int r = 0; r < 8; ++r) {
          float gi = acc4[0][r] * 0.0625f + bfr[cl][0];
          float gf = acc4[1][r] * 0.0625f + bfr[cl][1];
          float gg = acc4[2][r] * 0.0625f + bfr[cl][2];
          float go = acc4[3][r] * 0.0625f + bfr[cl][3];
          float i_ = fast_sigmoid(gi);
          float f_ = fast_sigmoid(gf);
          float g_ = fast_tanh(gg);
          float o_ = fast_sigmoid(go);
          float c  = f_ * creg[mt][cl][r] + i_ * g_;
          creg[mt][cl][r] = c;
          float h = o_ * fast_tanh(c);
          hv[r] = h;
          hf[(mt * 16 + r + 8 * lh) * SF + j] = h;
        }
#if defined(__gfx1250__) && __has_builtin(__builtin_amdgcn_cvt_pk_fp8_f32)
#pragma unroll
        for (int r = 0; r < 8; r += 2) {
          int pk = __builtin_amdgcn_cvt_pk_fp8_f32(hv[r], hv[r + 1], 0, false);
          int m2 = mt * 16 + r + 8 * lh;
          hnxt[m2 * S8 + j]       = (unsigned char)(pk & 0xff);
          hnxt[(m2 + 1) * S8 + j] = (unsigned char)((pk >> 8) & 0xff);
        }
#else
#pragma unroll
        for (int r = 0; r < 8; ++r)
          hnxt[(mt * 16 + r + 8 * lh) * S8 + j] = f32_to_e4m3(hv[r]);
#endif
      }
    }
    __syncthreads();

    if (tid < 2 * BM) {
      int m = tid >> 1, o = tid & 1;
      const float4* hp  = (const float4*)(hf + m * SF);
      const float4* wp4 = (const float4*)(fcw_s + o * 256);
      float acc = 0.f;
#pragma unroll 8
      for (int jj = 0; jj < 64; ++jj) {
        float4 h4 = hp[jj], w4 = wp4[jj];
        acc += h4.x * w4.x + h4.y * w4.y + h4.z * w4.z + h4.w * w4.w;
      }
      out[(size_t)(rowbase + m) * (T * 2) + t * 2 + o] = acc + ((o == 0) ? fcb0 : fcb1);
    }
    __syncthreads();
  }
}

extern "C" void kernel_launch(void* const* d_in, const int* in_sizes, int n_in,
                              void* d_out, int out_size, void* d_ws, size_t ws_size,
                              hipStream_t stream) {
  const float* h   = (const float*)d_in[0];
  const float* c   = (const float*)d_in[1];
  const float* x0  = (const float*)d_in[2];
  const float* Wih = (const float*)d_in[3];
  const float* Whh = (const float*)d_in[4];
  const float* bih = (const float*)d_in[5];
  const float* bhh = (const float*)d_in[6];
  const float* fcW = (const float*)d_in[7];
  const float* fcb = (const float*)d_in[8];
  const int*   Tp  = (const int*)d_in[9];
  float* out = (float*)d_out;
  unsigned char* wfrag = (unsigned char*)d_ws;

  const int B = in_sizes[0] / 256;
  lstm_prep<<<(64 * 2 * 32 * 64 + 255) / 256, 256, 0, stream>>>(Wih, Whh, fcW, wfrag);
  lstm_decode<<<B / BM, 256, SMEM_BYTES, stream>>>(h, c, x0, Wih, bih, bhh, fcW, fcb,
                                                   wfrag, Tp, out);
}
